// FaustVertexClassifier_57578331570653
// MI455X (gfx1250) — matrix operand order repaired and checked by reading, not compiled, not run on hardware
//
#include <hip/hip_runtime.h>
#include <hip/hip_bf16.h>

// ---------------------------------------------------------------------------
// FAUST vertex classifier: 3x (barycentric interp -> rotated template GEMM ->
// bias+relu -> angular max) -> dense classifier.
// GEMM: v_wmma_f32_16x16x32_bf16 with B staged via TDM (tensor_load_to_lds,
// double-buffered LDS) and per-wave A fragments from global.
// ---------------------------------------------------------------------------

typedef __attribute__((ext_vector_type(16))) __bf16 v16bf;
typedef __attribute__((ext_vector_type(8)))  __bf16 bf16x8;
typedef __attribute__((ext_vector_type(8)))  float  v8f;
typedef __attribute__((ext_vector_type(4)))  unsigned int v4u;
typedef __attribute__((ext_vector_type(8)))  int v8i;
typedef __attribute__((ext_vector_type(4)))  int v4i;

#define NV 6890     // vertices (== n_classes)
#define NR 5
#define NA 8
#define RA (NR*NA)  // 40
#define KC 64       // K-chunk staged per TDM op (all layer K are multiples)

__device__ __forceinline__ __bf16 f2bf(float f) {
  unsigned u = __builtin_bit_cast(unsigned, f);
  unsigned r = (u + 0x7FFFu + ((u >> 16) & 1u)) >> 16;
  unsigned short s = (unsigned short)r;
  return __builtin_bit_cast(__bf16, s);
}

// --- WMMA A-fragment (16x32 bf16 row-major, wave32, this code's contiguous k order) ------------
// lane l: row = l&15 ; kbase = (l>>4)*16 ; vgpr0-3 = k[kbase..+7], vgpr4-7 =
// k[kbase+8..+15]
__device__ __forceinline__ v16bf load_a_frag(const __bf16* __restrict__ rowk, int lane) {
  const __bf16* p = rowk + ((lane >> 4) << 4);
  bf16x8 lo = *reinterpret_cast<const bf16x8*>(p);
  bf16x8 hi = *reinterpret_cast<const bf16x8*>(p + 8);
  v16bf f;
#pragma unroll
  for (int i = 0; i < 8; ++i) { f[i] = lo[i]; f[8 + i] = hi[i]; }
  return f;
}
// B-fragment from t-major memory (for the dense head): lane l: col = l&15,
// kbase = (l>>4)*16, 16 contiguous k per lane half.
__device__ __forceinline__ v16bf load_b_frag(const __bf16* __restrict__ colk, int lane) {
  const __bf16* p = colk + ((lane >> 4) << 4);
  bf16x8 lo = *reinterpret_cast<const bf16x8*>(p);
  bf16x8 hi = *reinterpret_cast<const bf16x8*>(p + 8);
  v16bf f;
#pragma unroll
  for (int i = 0; i < 8; ++i) { f[i] = lo[i]; f[8 + i] = hi[i]; }
  return f;
}

// --- 1) barycentric interpolation: sig[V,C] -> bf16 interp[V, RA*C] --------
__global__ void interp_kernel(const float* __restrict__ sig,
                              const int*   __restrict__ idx,
                              const float* __restrict__ wgt,
                              __bf16* __restrict__ out, int C) {
  const int v = blockIdx.x, ra = blockIdx.y;
  const long b3 = ((long)v * RA + ra) * 3;
  const int  i0 = idx[b3], i1 = idx[b3 + 1], i2 = idx[b3 + 2];
  const float w0 = wgt[b3], w1 = wgt[b3 + 1], w2 = wgt[b3 + 2];
  const float* s0 = sig + (long)i0 * C;
  const float* s1 = sig + (long)i1 * C;
  const float* s2 = sig + (long)i2 * C;
  __bf16* o = out + ((long)v * RA + ra) * C;
  for (int c = threadIdx.x; c < C; c += blockDim.x)
    o[c] = f2bf(w0 * s0[c] + w1 * s1[c] + w2 * s2[c]);
}

// --- 2) rotated templates: T[t,r,a,c] -> bf16 Bt[o][t][(r*8+a)*C + c] ------
__global__ void rotb_kernel(const float* __restrict__ Tp,
                            __bf16* __restrict__ Bt, int Tdim, int C) {
  const int t = blockIdx.x, o = blockIdx.y;
  const int K = RA * C;
  const float* src = Tp + (long)t * K;
  __bf16* dst = Bt + ((long)o * Tdim + t) * K;
  for (int k = threadIdx.x; k < K; k += blockDim.x) {
    int c = k % C, ra = k / C;
    int a = ra & 7, r = ra >> 3;
    int asrc = (a - o) & 7;              // B_o[(r,a,c)][t] = T[t,r,(a-o)%8,c]
    dst[k] = f2bf(src[((r << 3) + asrc) * C + c]);
  }
}

// --- TDM descriptor: 3D tile  x=KC (k), y=16 (t rows), z=8 (rotations) -----
// Loads 8*16*KC bf16 contiguously into LDS at lds_off:
//   LDS[((z*16 + y)*KC + x)*2] = MEM[gaddr + 2*(x + y*K + z*zstride)]
__device__ __forceinline__ void tdm_load_b(unsigned lds_off,
                                           unsigned long long gaddr,
                                           unsigned K, unsigned long long zstride) {
  v4u g0;
  g0[0] = 1u;                                   // count=1, no gather/restore
  g0[1] = lds_off;                              // lds_addr [63:32]
  g0[2] = (unsigned)(gaddr & 0xFFFFFFFFull);    // global_addr [95:64]
  g0[3] = (unsigned)((gaddr >> 32) & 0x1FFFFFFull) | (2u << 30); // [120:96]+type=2
  v8i g1;
  g1[0] = (int)(1u << 16);                      // data_size=1 (2 bytes)
  g1[1] = (int)((unsigned)(KC & 0xFFFFu) << 16);          // tensor_dim0 lo16
  g1[2] = (int)(((unsigned)KC >> 16) | (16u << 16));      // dim0 hi16 | dim1 lo16
  g1[3] = (int)((16u >> 16) | ((unsigned)KC << 16));      // dim1 hi16 | tile_dim0
  g1[4] = (int)(16u | (8u << 16));              // tile_dim1=16, tile_dim2=8
  g1[5] = (int)K;                               // tensor_dim0_stride lo32
  g1[6] = (int)((K >> 31 >> 1) | ((unsigned)(zstride & 0xFFFFull) << 16)); // stride0 hi16 | stride1 lo16
  g1[7] = (int)((zstride >> 16) & 0xFFFFFFFFull);         // tensor_dim1_stride hi32
  v4i g2;
  g2[0] = 8;                                    // tensor_dim2 = 8 (z extent)
  g2[1] = 0; g2[2] = 0; g2[3] = 0;
  v4i g3; g3[0] = 0; g3[1] = 0; g3[2] = 0; g3[3] = 0;
  v8i g4;                                       // extra group (clang-23 form)
#pragma unroll
  for (int i = 0; i < 8; ++i) g4[i] = 0;
  __builtin_amdgcn_tensor_load_to_lds(g0, g1, g2, g3, g4, 0);
}

// --- 3) rotated-template GEMM + bias + relu + angular max ------------------
// X[v,t] = relu( max_o  sum_k A[v,k] * Bt[o][t][k]  + bias[t] )
// Block: 8 waves, one shared ntile, 8 consecutive mtiles.  B staged by TDM
// into double-buffered LDS; A fragments per-wave from global.
__global__ __launch_bounds__(256) void gemm_rot_kernel(
    const __bf16* __restrict__ A, const __bf16* __restrict__ Bt,
    const float* __restrict__ bias, float* __restrict__ X,
    int Tdim, int K, int NT, int MT) {
  __shared__ __align__(16) __bf16 smem[2][8 * 16 * KC];   // 2 x 16KB

  const int wid = threadIdx.x >> 5, lane = threadIdx.x & 31;
  const int ntile = blockIdx.x % NT;
  int mtile = (blockIdx.x / NT) * 8 + wid;                 // wave-uniform
  if (mtile >= MT) mtile = MT - 1;   // keep wave alive for barriers (dup tile)

  int arow = mtile * 16 + (lane & 15); if (arow > NV - 1) arow = NV - 1;
  const __bf16* ap = A + (long)arow * K;

  // B tile base for this ntile (rotation handled by TDM z-dimension)
  const unsigned long long bbase =
      (unsigned long long)(Bt + (long)(ntile * 16) * K);
  const unsigned long long zstride = (unsigned long long)Tdim * (unsigned)K;

  v8f acc[8];
#pragma unroll
  for (int o = 0; o < 8; ++o)
#pragma unroll
    for (int j = 0; j < 8; ++j) acc[o][j] = 0.0f;

  const int nchunk = K / KC;
  // prologue: stage chunk 0 into buffer 0
  if (wid == 0)
    tdm_load_b((unsigned)(size_t)&smem[0][0], bbase, (unsigned)K, zstride);

  for (int ch = 0; ch < nchunk; ++ch) {
    const int kk = ch * KC;
    // private A fragments for this chunk (global loads overlap the barrier)
    v16bf a0 = load_a_frag(ap + kk, lane);
    v16bf a1 = load_a_frag(ap + kk + 32, lane);

    if (wid == 0) {
      if (ch + 1 < nchunk) {
        tdm_load_b((unsigned)(size_t)&smem[(ch + 1) & 1][0],
                   bbase + (unsigned long long)(kk + KC) * 2,
                   (unsigned)K, zstride);
        __builtin_amdgcn_s_wait_tensorcnt(1);   // chunk ch landed
      } else {
        __builtin_amdgcn_s_wait_tensorcnt(0);
      }
    }
    __syncthreads();                            // publish buffer ch&1

    const __bf16* sb = &smem[ch & 1][0];
#pragma unroll
    for (int s = 0; s < 2; ++s) {
      v16bf bf[8];
#pragma unroll
      for (int o = 0; o < 8; ++o) {
        const __bf16* q = sb + ((o * 16 + (lane & 15)) * KC + s * 32 +
                                ((lane >> 4) << 4));
        bf16x8 lo = *reinterpret_cast<const bf16x8*>(q);
        bf16x8 hi = *reinterpret_cast<const bf16x8*>(q + 8);
#pragma unroll
        for (int i = 0; i < 8; ++i) { bf[o][i] = lo[i]; bf[o][8 + i] = hi[i]; }
      }
      const v16bf& a = s ? a1 : a0;
#pragma unroll
      for (int o = 0; o < 8; ++o)
        acc[o] = __builtin_amdgcn_wmma_f32_16x16x32_bf16(
            false, a, false, bf[o], (short)0, acc[o], false, false);
    }
    __syncthreads();                            // done reading buffer ch&1
  }

  const int n = ntile * 16 + (lane & 15);
  const float bn = bias[n];
  const int mbase = mtile * 16 + ((lane >> 4) << 3);
#pragma unroll
  for (int i = 0; i < 8; ++i) {
    float r = acc[0][i];
#pragma unroll
    for (int o = 1; o < 8; ++o) r = fmaxf(r, acc[o][i]);
    r = fmaxf(r + bn, 0.0f);                    // relu after angular max
    const int m = mbase + i;
    if (m < NV) X[(long)m * Tdim + n] = r;
  }
}

// --- 4) misc converters ----------------------------------------------------
__global__ void cvtbf_kernel(const float* __restrict__ in,
                             __bf16* __restrict__ out, int n) {
  int i = blockIdx.x * blockDim.x + threadIdx.x;
  if (i < n) out[i] = f2bf(in[i]);
}
__global__ void wdt_kernel(const float* __restrict__ Wd,
                           __bf16* __restrict__ Wdt) {
  const int n = blockIdx.x;                     // Wd[128, V] -> Wdt[V,128]
  for (int k = threadIdx.x; k < 128; k += blockDim.x)
    Wdt[(long)n * 128 + k] = f2bf(Wd[(long)k * NV + n]);
}

// --- 5) dense classifier: out[v,n] = X[v,:128] . Wd[:,n] + bd[n] -----------
__global__ __launch_bounds__(256) void dense_kernel(
    const __bf16* __restrict__ A, const __bf16* __restrict__ Bt,
    const float* __restrict__ bd, float* __restrict__ out, int NT, int MT) {
  const int wid = threadIdx.x >> 5, lane = threadIdx.x & 31;
  const int ntile = blockIdx.x % NT;
  const int mtile = (blockIdx.x / NT) * 8 + wid;
  if (mtile >= MT) return;                      // no barriers below: safe
  const int Kd = 128;
  int arow = mtile * 16 + (lane & 15); if (arow > NV - 1) arow = NV - 1;
  int brow = ntile * 16 + (lane & 15); if (brow > NV - 1) brow = NV - 1;
  const __bf16* ap = A  + (long)arow * Kd;
  const __bf16* bp = Bt + (long)brow * Kd;

  v8f acc;
#pragma unroll
  for (int j = 0; j < 8; ++j) acc[j] = 0.0f;
#pragma unroll
  for (int kk = 0; kk < 128; kk += 32) {
    v16bf a = load_a_frag(ap + kk, lane);
    v16bf b = load_b_frag(bp + kk, lane);
    acc = __builtin_amdgcn_wmma_f32_16x16x32_bf16(
        false, a, false, b, (short)0, acc, false, false);
  }
  const int n = ntile * 16 + (lane & 15);
  int nc = n > NV - 1 ? NV - 1 : n;
  const float bn = bd[nc];
  const int mbase = mtile * 16 + ((lane >> 4) << 3);
#pragma unroll
  for (int i = 0; i < 8; ++i) {
    const int m = mbase + i;
    if (m < NV && n < NV) out[(long)m * NV + n] = acc[i] + bn;
  }
}

// ---------------------------------------------------------------------------
extern "C" void kernel_launch(void* const* d_in, const int* in_sizes, int n_in,
                              void* d_out, int out_size, void* d_ws, size_t ws_size,
                              hipStream_t stream) {
  const float* signal = (const float*)d_in[0];   // [V, 544]
  const int*   bcidx  = (const int*)  d_in[1];   // [V, 5, 8, 3]
  const float* bcw    = (const float*)d_in[2];   // [V, 5, 8, 3]
  const float* t0 = (const float*)d_in[3]; const float* b0 = (const float*)d_in[4];
  const float* t1 = (const float*)d_in[5]; const float* b1 = (const float*)d_in[6];
  const float* t2 = (const float*)d_in[7]; const float* b2 = (const float*)d_in[8];
  const float* Wd = (const float*)d_in[9]; const float* bd = (const float*)d_in[10];
  float* out = (float*)d_out;

  // workspace carve-out
  size_t off = 0;
  auto carve = [&](size_t bytes) -> void* {
    void* p = (char*)d_ws + off;
    off += (bytes + 255) & ~(size_t)255;
    return p;
  };
  __bf16* bufA = (__bf16*)carve((size_t)NV * 21760 * 2);     // interp (max K)
  __bf16* bufB = (__bf16*)carve((size_t)8 * 21760 * 96 * 2); // rotated B (max)
  float*  X0   = (float*)carve((size_t)NV * 128 * 4);
  float*  X1   = (float*)carve((size_t)NV * 128 * 4);
  __bf16* Xb   = (__bf16*)carve((size_t)NV * 128 * 2);
  __bf16* Wdt  = (__bf16*)carve((size_t)NV * 128 * 2);

  const int MT = (NV + 15) / 16;          // 431
  const int MB = (MT + 7) / 8;            // 54 m-tile groups per n-tile

  struct Layer { const float* sig; const float* T; const float* b;
                 float* X; int C; int Tdim; };
  Layer layers[3] = {
    { signal, t0, b0, X0, 544,  96 },
    { X0,     t1, b1, X1,  96, 128 },
    { X1,     t2, b2, X0, 128, 128 },
  };

  for (int l = 0; l < 3; ++l) {
    const Layer& L = layers[l];
    const int K  = RA * L.C;                // 21760 / 3840 / 5120 (÷64 exact)
    const int NT = L.Tdim / 16;
    interp_kernel<<<dim3(NV, RA), 128, 0, stream>>>(L.sig, bcidx, bcw, bufA, L.C);
    rotb_kernel<<<dim3(L.Tdim, 8), 128, 0, stream>>>(L.T, bufB, L.Tdim, L.C);
    gemm_rot_kernel<<<NT * MB, 256, 0, stream>>>(bufA, bufB, L.b, L.X,
                                                 L.Tdim, K, NT, MT);
  }

  // dense head
  cvtbf_kernel<<<(NV * 128 + 255) / 256, 256, 0, stream>>>(X0, Xb, NV * 128);
  wdt_kernel<<<NV, 128, 0, stream>>>(Wd, Wdt);
  dense_kernel<<<MT * MB, 256, 0, stream>>>(Xb, Wdt, bd, out, MT, MT);
}
